// SpikingLinear_66898410602864
// MI455X (gfx1250) — matrix operand order repaired and checked by reading, not compiled, not run on hardware
//
#include <hip/hip_runtime.h>

// ---------------------------------------------------------------------------
// Spiking linear layer (LIF) for MI455X / gfx1250.
//   proj[t] = input[t] @ W^T  (t = 0..98)  -- v_wmma_f32_16x16x32_f16
//   LIF scan over t                        -- elementwise, memory bound
// Fast path: pre-convert input/weight to padded f16 in d_ws, GEMM loads f16
// directly (no cvt in the hot loop), 2x2 register tiling (32x32 per wave).
// Fallback (ws too small): convert f32->f16 on the fly inside the GEMM.
// proj is staged in-place in d_out slot t+1, then overwritten by spikes.
// ---------------------------------------------------------------------------

#define STEPS   100
#define BATCH   256
#define D1      784
#define D2      400
#define KP      800     // K padded to a multiple of 32

typedef __attribute__((ext_vector_type(16))) _Float16 v16h;
typedef __attribute__((ext_vector_type(8)))  _Float16 v8h;
typedef __attribute__((ext_vector_type(8)))  float    v8f;

constexpr int M_TOTAL = (STEPS - 1) * BATCH;   // 25344 rows (t*256 + b)
constexpr int MT      = M_TOTAL / 16;          // 1584 M tiles
constexpr int NT      = D2 / 16;               // 25  N tiles
constexpr int MP      = MT / 2;                // 792 M tile-pairs
constexpr int NP      = (NT + 1) / 2;          // 13  N tile-pairs (last one dup)
constexpr int KCH     = KP / 8;                // 100 8-half chunks per row

__device__ __forceinline__ v8f wmma_f16(const v16h& a, const v16h& b, const v8f& c) {
    return __builtin_amdgcn_wmma_f32_16x16x32_f16(false, a, false, b, (short)0, c,
                                                  false, false);
}

// ---------------------------------------------------------------------------
// f32 -> f16 conversion with K padded 784 -> 800 (pad = 0). One thread per
// 8-half chunk; 784 % 8 == 0 so chunks are either fully valid or fully pad.
// ---------------------------------------------------------------------------
__global__ void __launch_bounds__(256)
convert_f16_pad(const float* __restrict__ src, _Float16* __restrict__ dst, int rows)
{
    const int idx   = blockIdx.x * 256 + threadIdx.x;
    const int total = rows * KCH;
    if (idx >= total) return;
    const int r = idx / KCH;
    const int k = (idx - r * KCH) * 8;
    v8h v = {};
    if (k < D1) {
        const float* s = src + (size_t)r * D1 + k;
#pragma unroll
        for (int i = 0; i < 8; ++i) v[i] = (_Float16)s[i];
    }
    *(v8h*)(dst + (size_t)r * KP + k) = v;
}

// ---------------------------------------------------------------------------
// Fast GEMM: f16 operands, 2x2 tiles (32x32 output) per wave32.
// ---------------------------------------------------------------------------
__global__ void __launch_bounds__(256)
snn_gemm_wmma_f16(const _Float16* __restrict__ a16,   // (25344, 800) f16
                  const _Float16* __restrict__ b16,   // (400,   800) f16
                  float* __restrict__ out)            // (100,256,400) f32
{
    const int wave = blockIdx.x * 8 + (threadIdx.x >> 5);   // 10296 total, exact
    const int mp   = wave / NP;
    const int np   = wave - mp * NP;

    const int lane = threadIdx.x & 31;
    const int hs   = lane >> 4;
    const int l    = lane & 15;

    const int n0 = np * 2;
    const int n1 = (np * 2 + 1 < NT) ? (np * 2 + 1) : (NT - 1);  // dup on last pair

    const _Float16* __restrict__ arow0 = a16 + (size_t)(mp * 32 + l) * KP;
    const _Float16* __restrict__ arow1 = arow0 + (size_t)16 * KP;
    const _Float16* __restrict__ brow0 = b16 + (size_t)(n0 * 16 + l) * KP;
    const _Float16* __restrict__ brow1 = b16 + (size_t)(n1 * 16 + l) * KP;

    v8f acc00 = {}, acc01 = {}, acc10 = {}, acc11 = {};

#pragma unroll 2
    for (int k0 = 0; k0 < KP; k0 += 32) {
        const int ao = k0 + 16 * hs;       // A halves 0..15 contiguous at ao
        const int bo = k0 + 16 * hs;       // B halves 0..15 contiguous at bo

        const v8h a0lo = *(const v8h*)(arow0 + ao);
        const v8h a0hi = *(const v8h*)(arow0 + ao + 8);
        const v8h a1lo = *(const v8h*)(arow1 + ao);
        const v8h a1hi = *(const v8h*)(arow1 + ao + 8);
        const v16h b0  = *(const v16h*)(brow0 + bo);
        const v16h b1  = *(const v16h*)(brow1 + bo);

        v16h a0, a1;
#pragma unroll
        for (int i = 0; i < 8; ++i) {
            a0[i] = a0lo[i]; a0[i + 8] = a0hi[i];
            a1[i] = a1lo[i]; a1[i + 8] = a1hi[i];
        }

        acc00 = wmma_f16(a0, b0, acc00);
        acc01 = wmma_f16(a0, b1, acc01);
        acc10 = wmma_f16(a1, b0, acc10);
        acc11 = wmma_f16(a1, b1, acc11);
    }

    // Store. D VGPR r holds (M = r + 8*hs, N = l) of each 16x16 tile.
    const int m_base = mp * 32;            // 32 | 256 -> single t per wave
    const int t      = m_base >> 8;
    const int b_base = m_base & 255;
    float* obase = out + (size_t)(t + 1) * (BATCH * D2);

    auto store_tile = [&](const v8f& acc, int mi, int nt) {
        float* ob = obase + (size_t)(b_base + mi * 16 + 8 * hs) * D2 + nt * 16 + l;
#pragma unroll
        for (int r = 0; r < 8; ++r) ob[(size_t)r * D2] = acc[r];
    };

    store_tile(acc00, 0, n0);
    store_tile(acc10, 1, n0);
    if (np * 2 + 1 < NT) {                 // uniform per wave
        store_tile(acc01, 0, n1);
        store_tile(acc11, 1, n1);
    }
}

// ---------------------------------------------------------------------------
// Fallback GEMM (no workspace): convert f32->f16 on the fly, 16x16 per wave.
// ---------------------------------------------------------------------------
__global__ void __launch_bounds__(256)
snn_gemm_wmma_f32src(const float* __restrict__ inp,
                     const float* __restrict__ wgt,
                     float* __restrict__ out)
{
    const int wave   = blockIdx.x * 8 + (threadIdx.x >> 5);  // 39600 total, exact
    const int m_tile = wave / NT;
    const int n_tile = wave - m_tile * NT;

    const int lane = threadIdx.x & 31;
    const int hs   = lane >> 4;
    const int l    = lane & 15;

    const float* __restrict__ arow = inp + (size_t)(m_tile * 16 + l) * D1;
    const float* __restrict__ brow = wgt + (size_t)(n_tile * 16 + l) * D1;

    v8f acc = {};
    int k0 = 0;
#pragma unroll 2
    for (int kk = 0; kk < D1 / 32; ++kk, k0 += 32) {
        v16h a, b;
        const float* ap0 = arow + k0 + 8 * hs;
        const float* ap1 = ap0 + 16;
#pragma unroll
        for (int i = 0; i < 8; ++i) { a[i] = (_Float16)ap0[i]; a[i + 8] = (_Float16)ap1[i]; }
        const float* bp = brow + k0 + 8 * hs;
#pragma unroll
        for (int i = 0; i < 8; ++i) { b[i] = (_Float16)bp[i]; b[i + 8] = (_Float16)bp[i + 16]; }
        acc = wmma_f16(a, b, acc);
    }
    {   // K tail: 768..783 valid, 784..799 zero
        v16h a = {}, b = {};
        const float* ap0 = arow + k0 + 8 * hs;
#pragma unroll
        for (int i = 0; i < 8; ++i) a[i] = (_Float16)ap0[i];
        {
            const float* bp = brow + k0 + 8 * hs;
#pragma unroll
            for (int i = 0; i < 8; ++i) b[i] = (_Float16)bp[i];
        }
        acc = wmma_f16(a, b, acc);
    }

    const int m_base = m_tile * 16;
    const int t      = m_base >> 8;
    const int b_base = m_base & 255;
    float* ob = out + (size_t)(t + 1) * (BATCH * D2)
                    + (size_t)(b_base + 8 * hs) * D2 + n_tile * 16 + l;
#pragma unroll
    for (int r = 0; r < 8; ++r) ob[(size_t)r * D2] = acc[r];
}

// ---------------------------------------------------------------------------
// LIF scan: one thread per 4 consecutive (b,o) states, float4 in-place update.
// ---------------------------------------------------------------------------
__global__ void __launch_bounds__(256)
snn_scan4(float* __restrict__ out)
{
    const int S4  = (BATCH * D2) / 4;          // 25600
    const int tid = blockIdx.x * blockDim.x + threadIdx.x;
    if (tid >= S4) return;

    const float a_m = 1.0f - 1.0f / 20.0f;
    const float b_m = 1.0f / 20.0f;
    const float a_s = 1.0f - 1.0f / 5.0f;

    float4* p = (float4*)out + tid;
    *p = make_float4(0.f, 0.f, 0.f, 0.f);      // t = 0 output is zeros
    p += S4;

    float V[4] = {0.f, 0.f, 0.f, 0.f};
    float I[4] = {0.f, 0.f, 0.f, 0.f};
    for (int t = 1; t < STEPS; ++t, p += S4) {
        float4 pr = *p;
        float  q[4] = {pr.x, pr.y, pr.z, pr.w};
        float  s[4];
#pragma unroll
        for (int j = 0; j < 4; ++j) {
            const float Vn = a_m * V[j] + b_m * I[j];
            I[j] = a_s * I[j] + q[j];
            s[j] = (Vn > 1.0f) ? 1.0f : 0.0f;
            V[j] = Vn - s[j] * Vn;
        }
        *p = make_float4(s[0], s[1], s[2], s[3]);
    }
}

// ---------------------------------------------------------------------------
extern "C" void kernel_launch(void* const* d_in, const int* in_sizes, int n_in,
                              void* d_out, int out_size, void* d_ws, size_t ws_size,
                              hipStream_t stream) {
    const float* inp = (const float*)d_in[0];   // (100,256,784) f32
    const float* wgt = (const float*)d_in[1];   // (400,784) f32
    float* out = (float*)d_out;                 // (100,256,400) f32

    const size_t a16_bytes = (size_t)M_TOTAL * KP * sizeof(_Float16); // 40,550,400
    const size_t b16_bytes = (size_t)D2 * KP * sizeof(_Float16);      //    640,000

    if (ws_size >= a16_bytes + b16_bytes) {
        _Float16* a16 = (_Float16*)d_ws;
        _Float16* b16 = (_Float16*)((char*)d_ws + a16_bytes);

        convert_f16_pad<<<dim3((M_TOTAL * KCH) / 256), dim3(256), 0, stream>>>(
            inp, a16, M_TOTAL);                                  // 9900 blocks exact
        convert_f16_pad<<<dim3((D2 * KCH + 255) / 256), dim3(256), 0, stream>>>(
            wgt, b16, D2);

        // 792*13 = 10296 waves / 8 per block = 1287 blocks exact
        snn_gemm_wmma_f16<<<dim3((MP * NP) / 8), dim3(256), 0, stream>>>(a16, b16, out);
    } else {
        // 1584*25 = 39600 waves / 8 per block = 4950 blocks exact
        snn_gemm_wmma_f32src<<<dim3((MT * NT) / 8), dim3(256), 0, stream>>>(inp, wgt, out);
    }

    const int S4 = (BATCH * D2) / 4;            // 25600 threads, 100 blocks
    snn_scan4<<<dim3(S4 / 256), dim3(256), 0, stream>>>(out);
}
